// LinearAttention_18829136625823
// MI455X (gfx1250) — matrix operand order repaired and checked by reading, not compiled, not run on hardware
//
#include <hip/hip_runtime.h>

// ---------------------------------------------------------------------------
// LinearAttention on MI455X (gfx1250): bf16 WMMA pipeline, LDS-staged GEMMs
//   B=4, N=8192, C=768, H=12, D=64
// ---------------------------------------------------------------------------

typedef __bf16 bf16;
typedef __attribute__((ext_vector_type(16))) __bf16 v16bf;
typedef __attribute__((ext_vector_type(8)))  __bf16 v8bf;
typedef __attribute__((ext_vector_type(8)))  float  v8f;

#define DEV static __device__ __forceinline__

static constexpr int   Bb = 4, Nn = 8192, Cc = 768, Hh = 12, Dd = 64;
static constexpr int   BH = Bb * Hh;              // 48
static constexpr float FEAT_SCALE = 0.35355339059327373f;  // 64^-0.25

// ---- async global->LDS copy (gfx1250), with compile-safe fallback ---------
#if defined(__has_builtin)
#if __has_builtin(__builtin_amdgcn_global_load_async_to_lds_b128)
#define HAS_ASYNC_LDS 1
#endif
#endif
#ifndef HAS_ASYNC_LDS
#define HAS_ASYNC_LDS 0
#endif

typedef int v4i_vs __attribute__((vector_size(16)));
typedef __attribute__((address_space(1))) v4i_vs* as1_v4i_p;   // global
typedef __attribute__((address_space(3))) v4i_vs* as3_v4i_p;   // LDS

DEV void cp_b128_to_lds(const bf16* g, bf16* l) {
#if HAS_ASYNC_LDS
    // AS1 bits == flat address; AS3 offset == low 32 bits of flat LDS address
    as1_v4i_p gp = (as1_v4i_p)(unsigned long long)g;
    as3_v4i_p lp = (as3_v4i_p)(unsigned int)(unsigned long long)l;
    __builtin_amdgcn_global_load_async_to_lds_b128(gp, lp, 0, 0);
#else
    *(v8bf*)l = *(const v8bf*)g;
#endif
}

DEV void async_copy_wait() {
#if HAS_ASYNC_LDS
#if __has_builtin(__builtin_amdgcn_s_wait_asynccnt)
    __builtin_amdgcn_s_wait_asynccnt(0);
#endif
#endif
}

// ---------------- WMMA fragment helpers (16x16x32 bf16) --------------------
DEV v16bf pack16(v8bf lo, v8bf hi) {
    v16bf f;
#pragma unroll
    for (int i = 0; i < 8; ++i) { f[i] = lo[i]; f[8 + i] = hi[i]; }
    return f;
}

// A (16x32, row-major source):
//   lanes 0-15 : row M=lane,    elems 0..15 = K+0..15
//   lanes16-31 : row M=lane-16, elems 0..15 = K+16..31
DEV v16bf load_frag_a(const bf16* __restrict__ A, int lda, int lane) {
    const int r = lane & 15;
    const int g = lane >> 4;
    const bf16* p = A + (size_t)r * lda + g * 16;
    return pack16(*(const v8bf*)p, *(const v8bf*)(p + 8));
}

// B (32x16) from B^T stored row-major (row = output column, contiguous K):
//   lanes 0-15 : col N=lane,    elems 0..15 = K+0..15
//   lanes16-31 : col N=lane-16, elems 0..15 = K+16..31
DEV v16bf load_frag_bt(const bf16* __restrict__ Bt, int ldb, int lane) {
    const int c = lane & 15;
    const int g = lane >> 4;
    const bf16* p = Bt + (size_t)c * ldb + g * 16;
    return pack16(*(const v8bf*)p, *(const v8bf*)(p + 8));
}

DEV v16bf load_frag_bt_f32(const float* __restrict__ Bt, int ldb, int lane) {
    const int c = lane & 15;
    const int g = lane >> 4;
    const float* p = Bt + (size_t)c * ldb + g * 16;
    v16bf f;
#pragma unroll
    for (int i = 0; i < 16; ++i) f[i] = (bf16)p[i];
    return f;
}

DEV v8f wmma_bf16(v16bf a, v16bf b, v8f c) {
    return __builtin_amdgcn_wmma_f32_16x16x32_bf16(
        false, a, false, b, (short)0, c, false, false);
}

// C/D layout: lane<16 -> col=lane rows 0..7 ; lane>=16 -> col=lane-16 rows 8..15

// ---------------------------------------------------------------------------
// Elementwise helpers
// ---------------------------------------------------------------------------
__global__ void k_f32_to_bf16(const float* __restrict__ in, bf16* __restrict__ out, int n) {
    int i = blockIdx.x * blockDim.x + threadIdx.x;
    if (i < n) out[i] = (bf16)in[i];
}

__global__ void k_zero_f32(float* __restrict__ p, int n) {
    int i = blockIdx.x * blockDim.x + threadIdx.x;
    if (i < n) p[i] = 0.0f;
}

// ---------------------------------------------------------------------------
// Tiled GEMM: D[M,N] = A(bf16)[M,K] * Bt(bf16)[N,K]^T  (+bias, f32 out if F32OUT)
//  - 256 threads / 8 waves per block, 128x128 output tile
//  - 128x32 A and Bt panels double-buffered in LDS via async copy
//  - wave computes 32(M) x 64(N): 8 accumulators, 8 WMMA per K-step
// ---------------------------------------------------------------------------
template <bool F32OUT>
__global__ __launch_bounds__(256)
void k_gemm_tile(const bf16* __restrict__ A, const bf16* __restrict__ Bt,
                 bf16* __restrict__ Dbf, float* __restrict__ Df32,
                 const float* __restrict__ bias,
                 int N, int K, int lda, int ldb, int ldd) {
    __shared__ __align__(16) bf16 As[2][128][40];   // 32 used + pad
    __shared__ __align__(16) bf16 Bs[2][128][40];

    const int nblk = N >> 7;
    const int m0 = (blockIdx.x / nblk) * 128;
    const int n0 = (blockIdx.x % nblk) * 128;
    const int tid  = threadIdx.x;
    const int wave = tid >> 5, lane = tid & 31;
    const int wm = wave >> 1;                 // 0..3 : 32-row strip
    const int wn = wave & 1;                  // 0..1 : 64-col strip
    const int srow = tid >> 1;                // 0..127 staging row
    const int sseg = (tid & 1) * 16;          // 0 or 16 (elements)

    const bf16* gA = A  + (size_t)(m0 + srow) * lda + sseg;
    const bf16* gB = Bt + (size_t)(n0 + srow) * ldb + sseg;

    auto stage = [&](int buf, int k0) {
        cp_b128_to_lds(gA + k0,     &As[buf][srow][sseg]);
        cp_b128_to_lds(gA + k0 + 8, &As[buf][srow][sseg + 8]);
        cp_b128_to_lds(gB + k0,     &Bs[buf][srow][sseg]);
        cp_b128_to_lds(gB + k0 + 8, &Bs[buf][srow][sseg + 8]);
    };

    v8f acc[2][4] = {};

    stage(0, 0);
    async_copy_wait();
    __syncthreads();

    const int KT = K >> 5;
    const int r = lane & 15, g = lane >> 4;
    for (int kt = 0; kt < KT; ++kt) {
        const int buf = kt & 1;
        if (kt + 1 < KT) stage(buf ^ 1, (kt + 1) * 32);

        v16bf afrag[2];
#pragma unroll
        for (int ms = 0; ms < 2; ++ms) {
            const bf16* p = &As[buf][wm * 32 + ms * 16 + r][g * 16];
            afrag[ms] = pack16(*(const v8bf*)p, *(const v8bf*)(p + 8));
        }
#pragma unroll
        for (int j = 0; j < 4; ++j) {
            const bf16* p = &Bs[buf][wn * 64 + j * 16 + r][g * 16];
            v16bf bfrag = pack16(*(const v8bf*)p, *(const v8bf*)(p + 8));
            acc[0][j] = wmma_bf16(afrag[0], bfrag, acc[0][j]);
            acc[1][j] = wmma_bf16(afrag[1], bfrag, acc[1][j]);
        }
        async_copy_wait();
        __syncthreads();
    }

    const int c = lane & 15, mb = (lane >> 4) * 8;
#pragma unroll
    for (int ms = 0; ms < 2; ++ms) {
        const int mrow = m0 + wm * 32 + ms * 16 + mb;
#pragma unroll
        for (int j = 0; j < 4; ++j) {
            const int col = n0 + wn * 64 + j * 16 + c;
            if (F32OUT) {
                const float bv = bias[col];
#pragma unroll
                for (int i = 0; i < 8; ++i)
                    Df32[(size_t)(mrow + i) * ldd + col] = acc[ms][j][i] + bv;
            } else {
#pragma unroll
                for (int i = 0; i < 8; ++i)
                    Dbf[(size_t)(mrow + i) * ldd + col] = (bf16)acc[ms][j][i];
            }
        }
    }
}

// ---------------------------------------------------------------------------
// Feature map: q'/k' = relu(q|k @ Wfeat^T + bfeat) * FEAT_SCALE
//   qk==0 -> q' stored [bh][n][d] ; qk==1 -> k' stored transposed [bh][d][n]
// ---------------------------------------------------------------------------
__global__ void k_featmap(const bf16* __restrict__ qkv, const bf16* __restrict__ Wf,
                          const float* __restrict__ bias,
                          bf16* __restrict__ qp, bf16* __restrict__ kpT) {
    const int wave = blockIdx.x * 8 + (threadIdx.x >> 5);
    const int lane = threadIdx.x & 31;
    const int NT = Nn / 16;                       // 512
    const int qk  = wave / (BH * NT);
    const int rem = wave % (BH * NT);
    const int bh = rem / NT, nt = rem % NT;
    const int b = bh / Hh, h = bh % Hh;
    const int n0 = nt * 16;

    const bf16* Ab = qkv + ((size_t)(b * Nn + n0)) * (3 * Cc) + qk * Cc + h * Dd;

    v8f acc[4] = {};
#pragma unroll
    for (int kt = 0; kt < 2; ++kt) {
        v16bf a = load_frag_a(Ab + kt * 32, 3 * Cc, lane);
#pragma unroll
        for (int j = 0; j < 4; ++j) {
            v16bf bfrag = load_frag_bt(Wf + (size_t)(j * 16) * Dd + kt * 32, Dd, lane);
            acc[j] = wmma_bf16(a, bfrag, acc[j]);
        }
    }
    const int c = lane & 15, mb = (lane >> 4) * 8;
    if (qk == 0) {
#pragma unroll
        for (int j = 0; j < 4; ++j) {
            const float bv = bias[j * 16 + c];
#pragma unroll
            for (int i = 0; i < 8; ++i) {
                float v = acc[j][i] + bv;
                v = (v > 0.0f ? v : 0.0f) * FEAT_SCALE;
                qp[((size_t)bh * Nn + n0 + mb + i) * Dd + j * 16 + c] = (bf16)v;
            }
        }
    } else {
#pragma unroll
        for (int j = 0; j < 4; ++j) {
            const float bv = bias[j * 16 + c];
            v8bf o;
#pragma unroll
            for (int i = 0; i < 8; ++i) {
                float v = acc[j][i] + bv;
                v = (v > 0.0f ? v : 0.0f) * FEAT_SCALE;
                o[i] = (bf16)v;
            }
            *(v8bf*)(kpT + ((size_t)bh * Dd + j * 16 + c) * Nn + n0 + mb) = o;
        }
    }
}

// ---------------------------------------------------------------------------
// LDS tile transpose of v: qkv[.., 2C + h*64 + e] -> vT[bh][e][n]
// ---------------------------------------------------------------------------
__global__ void k_vtrans(const bf16* __restrict__ qkv, bf16* __restrict__ vT) {
    __shared__ bf16 tile[64][72];
    const int blk = blockIdx.x;                   // bh*128 + ntile
    const int bh = blk >> 7, nt = blk & 127;
    const int b = bh / Hh, h = bh % Hh;
    const int n0 = nt * 64;
    const int tid = threadIdx.x;
#pragma unroll
    for (int r = 0; r < 16; ++r) {
        int idx = r * 256 + tid;
        int nl = idx >> 6, e = idx & 63;
        tile[nl][e] = qkv[((size_t)(b * Nn + n0 + nl)) * (3 * Cc) + 2 * Cc + h * Dd + e];
    }
    __syncthreads();
#pragma unroll
    for (int r = 0; r < 16; ++r) {
        int idx = r * 256 + tid;
        int el = idx >> 6, nl = idx & 63;
        vT[((size_t)bh * Dd + el) * Nn + n0 + nl] = tile[nl][el];
    }
}

// ---------------------------------------------------------------------------
// context[d][e] = sum_n k'[n,d] v[n,e]   (A = k'^T, Bt = v^T; K=8192 split 8x)
// accumulated into ctxT[bh][e][d] (transposed) via global f32 atomics
// ---------------------------------------------------------------------------
__global__ void k_context(const bf16* __restrict__ kpT, const bf16* __restrict__ vT,
                          float* __restrict__ ctxT) {
    const int wave = blockIdx.x * 8 + (threadIdx.x >> 5);
    const int lane = threadIdx.x & 31;
    const int ks = wave & 7;
    const int mt = (wave >> 3) & 3;
    const int bh = wave >> 5;
    const int kbase = ks * 1024;

    const bf16* Ab = kpT + (size_t)bh * Dd * Nn + (size_t)(mt * 16) * Nn + kbase;
    const bf16* Bbp = vT + (size_t)bh * Dd * Nn + kbase;

    v8f acc[4] = {};
    for (int kt = 0; kt < 32; ++kt) {
        v16bf a = load_frag_a(Ab + kt * 32, Nn, lane);
#pragma unroll
        for (int j = 0; j < 4; ++j) {
            v16bf bfrag = load_frag_bt(Bbp + (size_t)(j * 16) * Nn + kt * 32, Nn, lane);
            acc[j] = wmma_bf16(a, bfrag, acc[j]);
        }
    }
    const int c = lane & 15, mb = (lane >> 4) * 8;
    float* base = ctxT + (size_t)bh * Dd * Dd;
#pragma unroll
    for (int j = 0; j < 4; ++j)
#pragma unroll
        for (int i = 0; i < 8; ++i)
            atomicAdd(&base[(j * 16 + c) * Dd + mt * 16 + mb + i], acc[j][i]);
}

// ---------------------------------------------------------------------------
// out_head = q' @ context ; written to attn[n][h*64+e] (bf16, [32768 x 768])
// ---------------------------------------------------------------------------
__global__ void k_attn_out(const bf16* __restrict__ qp, const float* __restrict__ ctxT,
                           bf16* __restrict__ attn) {
    const int wave = blockIdx.x * 8 + (threadIdx.x >> 5);
    const int lane = threadIdx.x & 31;
    const int NT = Nn / 16;                       // 512
    const int bh = wave / NT, nt = wave % NT;
    const int b = bh / Hh, h = bh % Hh;
    const int n0 = nt * 16;

    const bf16* Ab = qp + (size_t)bh * Nn * Dd + (size_t)n0 * Dd;
    const float* Cb = ctxT + (size_t)bh * Dd * Dd;

    v8f acc[4] = {};
#pragma unroll
    for (int kt = 0; kt < 2; ++kt) {
        v16bf a = load_frag_a(Ab + kt * 32, Dd, lane);
#pragma unroll
        for (int j = 0; j < 4; ++j) {
            v16bf bfrag = load_frag_bt_f32(Cb + (size_t)(j * 16) * Dd + kt * 32, Dd, lane);
            acc[j] = wmma_bf16(a, bfrag, acc[j]);
        }
    }
    const int c = lane & 15, mb = (lane >> 4) * 8;
#pragma unroll
    for (int j = 0; j < 4; ++j)
#pragma unroll
        for (int i = 0; i < 8; ++i)
            attn[((size_t)(b * Nn + n0 + mb + i)) * Cc + h * Dd + j * 16 + c] = (bf16)acc[j][i];
}

// ---------------------------------------------------------------------------
extern "C" void kernel_launch(void* const* d_in, const int* in_sizes, int n_in,
                              void* d_out, int out_size, void* d_ws, size_t ws_size,
                              hipStream_t stream) {
    const float* x     = (const float*)d_in[0];   // [B,N,C]
    const float* Wqkv  = (const float*)d_in[1];   // [3C,C]
    const float* Wfeat = (const float*)d_in[2];   // [D,D]
    const float* bfeat = (const float*)d_in[3];   // [D]
    const float* Wproj = (const float*)d_in[4];   // [C,C]
    const float* bproj = (const float*)d_in[5];   // [C]
    float* out = (float*)d_out;

    char* ws = (char*)d_ws;
    size_t off = 0;
    auto carve = [&](size_t bytes) -> void* {
        void* p = ws + off;
        off = (off + bytes + 255) & ~(size_t)255;
        return p;
    };
    const size_t MN = (size_t)Bb * Nn;            // 32768 rows
    bf16* xb     = (bf16*)carve(MN * Cc * 2);
    bf16* wqkvb  = (bf16*)carve((size_t)3 * Cc * Cc * 2);
    bf16* wfeatb = (bf16*)carve((size_t)Dd * Dd * 2);
    bf16* wprojb = (bf16*)carve((size_t)Cc * Cc * 2);
    bf16* qkvb   = (bf16*)carve(MN * 3 * Cc * 2);
    bf16* qp     = (bf16*)carve((size_t)BH * Nn * Dd * 2);
    bf16* kpT    = (bf16*)carve((size_t)BH * Dd * Nn * 2);
    bf16* vT     = (bf16*)carve((size_t)BH * Dd * Nn * 2);
    float* ctxT  = (float*)carve((size_t)BH * Dd * Dd * 4);
    bf16* attnb  = (bf16*)carve(MN * Cc * 2);

    // 1) fp32 -> bf16 conversions + context zero-init
    {
        int n;
        n = (int)(MN * Cc);
        k_f32_to_bf16<<<(n + 255) / 256, 256, 0, stream>>>(x, xb, n);
        n = 3 * Cc * Cc;
        k_f32_to_bf16<<<(n + 255) / 256, 256, 0, stream>>>(Wqkv, wqkvb, n);
        n = Dd * Dd;
        k_f32_to_bf16<<<(n + 255) / 256, 256, 0, stream>>>(Wfeat, wfeatb, n);
        n = Cc * Cc;
        k_f32_to_bf16<<<(n + 255) / 256, 256, 0, stream>>>(Wproj, wprojb, n);
        n = BH * Dd * Dd;
        k_zero_f32<<<(n + 255) / 256, 256, 0, stream>>>(ctxT, n);
    }

    // 2) QKV projection: [32768,768] x [768,2304], 128x128 tiles
    {
        int blocks = (int)(MN / 128) * ((3 * Cc) / 128);   // 256 * 18
        k_gemm_tile<false><<<blocks, 256, 0, stream>>>(
            xb, wqkvb, qkvb, nullptr, nullptr, 3 * Cc, Cc, Cc, Cc, 3 * Cc);
    }

    // 3) feature maps (q' row-major, k' transposed)
    k_featmap<<<(2 * BH * (Nn / 16)) / 8, 256, 0, stream>>>(qkvb, wfeatb, bfeat, qp, kpT);

    // 4) v transpose via LDS
    k_vtrans<<<BH * (Nn / 64), 256, 0, stream>>>(qkvb, vT);

    // 5) context = k'^T v  (K split 8x, f32 atomics into ctxT)
    k_context<<<(BH * 4 * 8) / 8, 256, 0, stream>>>(kpT, vT, ctxT);

    // 6) out_head = q' @ context
    k_attn_out<<<(BH * (Nn / 16)) / 8, 256, 0, stream>>>(qp, ctxT, attnb);

    // 7) final projection + bias -> d_out (f32), 128x128 tiles
    {
        int blocks = (int)(MN / 128) * (Cc / 128);         // 256 * 6
        k_gemm_tile<true><<<blocks, 256, 0, stream>>>(
            attnb, wprojb, nullptr, out, bproj, Cc, Cc, Cc, Cc, Cc);
    }
}
